// Int8BaichuanMLP_60413009986249
// MI455X (gfx1250) — 8-bit matrix operand order repaired and checked by reading, compiled, not run on hardware
//
#include <hip/hip_runtime.h>
#include <hip/hip_fp16.h>
#include <stdint.h>

#define M_TOK 8192
#define HDIM  4096
#define IDIM  11008

typedef __attribute__((ext_vector_type(8))) int v8i;

static __device__ __forceinline__ v8i wmma_iu8(v8i a, v8i b, v8i c) {
  return __builtin_amdgcn_wmma_i32_16x16x64_iu8(true, a, true, b, c, false, false);
}

static __device__ __forceinline__ void async_b128(uint32_t lds_addr, const void* gaddr) {
  asm volatile("global_load_async_to_lds_b128 %0, %1, off"
               :: "v"(lds_addr), "v"(gaddr) : "memory");
}

static __device__ __forceinline__ v8i load_a_frag(const int8_t* p) {
  v8i f;
  int2 t0 = *(const int2*)(p);
  int2 t1 = *(const int2*)(p + 8);
  int2 t2 = *(const int2*)(p + 32);
  int2 t3 = *(const int2*)(p + 40);
  f[0] = t0.x; f[1] = t0.y; f[2] = t1.x; f[3] = t1.y;
  f[4] = t2.x; f[5] = t2.y; f[6] = t3.x; f[7] = t3.y;
  return f;
}

static __device__ __forceinline__ v8i load_b_frag(const int8_t* p) {
  v8i f;
  int4 t0 = *(const int4*)(p);
  int4 t1 = *(const int4*)(p + 32);
  f[0] = t0.x; f[1] = t0.y; f[2] = t0.z; f[3] = t0.w;
  f[4] = t1.x; f[5] = t1.y; f[6] = t1.z; f[7] = t1.w;
  return f;
}

__global__ void pack_w_i8(const int* __restrict__ src, uint32_t* __restrict__ dst, int n4) {
  int t = blockIdx.x * blockDim.x + threadIdx.x;
  if (t >= n4) return;
  int4 v = ((const int4*)src)[t];
  dst[t] = (uint32_t)(v.x & 0xff) | ((uint32_t)(v.y & 0xff) << 8) |
           ((uint32_t)(v.z & 0xff) << 16) | ((uint32_t)(v.w & 0xff) << 24);
}

__global__ void quant_x_i8(const float* __restrict__ src, uint32_t* __restrict__ dst,
                           const float* __restrict__ scale, int n4) {
  int t = blockIdx.x * blockDim.x + threadIdx.x;
  if (t >= n4) return;
  float inv = 1.0f / scale[0];
  float4 v = ((const float4*)src)[t];
  int a = (int)rintf(v.x * inv); a = a < -127 ? -127 : (a > 127 ? 127 : a);
  int b = (int)rintf(v.y * inv); b = b < -127 ? -127 : (b > 127 ? 127 : b);
  int c = (int)rintf(v.z * inv); c = c < -127 ? -127 : (c > 127 ? 127 : c);
  int d = (int)rintf(v.w * inv); d = d < -127 ? -127 : (d > 127 ? 127 : d);
  dst[t] = (uint32_t)(a & 0xff) | ((uint32_t)(b & 0xff) << 8) |
           ((uint32_t)(c & 0xff) << 16) | ((uint32_t)(d & 0xff) << 24);
}

__global__ void __launch_bounds__(512)
gemm1_gate_up_silu(const int8_t* __restrict__ Aq,
                   const int8_t* __restrict__ Wg,
                   const int8_t* __restrict__ Wu,
                   const float* __restrict__ bias_g,
                   const float* __restrict__ bias_u,
                   const float* __restrict__ s_in,
                   const float* __restrict__ s_wg,
                   const float* __restrict__ s_wu,
                   const float* __restrict__ s_dn,
                   int8_t* __restrict__ Hq)
{
  const int K = HDIM;
  __shared__ int8_t lds[2][24576];

  const int t    = threadIdx.x;
  const int wave = t >> 5;
  const int lane = t & 31;
  const int wm = wave & 3, wn = wave >> 2;
  const int block_m = blockIdx.x * 128;
  const int block_n = blockIdx.y * 128;
  const int mr = block_m + wm * 32;
  const int nb = block_n + wn * 32;
  const int r = lane & 15, h = lane >> 4;

  const int row  = t >> 2;
  const int koff = (t & 3) * 16;
  const int8_t* gA  = Aq + (size_t)(block_m + row) * K + koff;
  const int8_t* gBg = Wg + (size_t)(block_n + row) * K + koff;
  const int8_t* gBu = Wu + (size_t)(block_n + row) * K + koff;
  const uint32_t ldsBase = (uint32_t)(uintptr_t)&lds[0][0];
  const uint32_t cA  = ldsBase + (uint32_t)(t * 16);
  const uint32_t cBg = cA + 8192;
  const uint32_t cBu = cA + 16384;

  auto issue = [&](uint32_t bufbase, int kb) {
    async_b128(cA + bufbase, gA + kb);
    async_b128(cBg + bufbase, gBg + kb);
    async_b128(cBu + bufbase, gBu + kb);
  };

  const uint32_t aOff0 = (uint32_t)((wm * 32 + r) * 64 + h * 16);
  const uint32_t aOff1 = aOff0 + 1024;
  const uint32_t gOff0 = (uint32_t)(8192 + (wn * 32 + r) * 64 + h * 16);
  const uint32_t gOff1 = gOff0 + 1024;
  const uint32_t uOff0 = gOff0 + 8192;
  const uint32_t uOff1 = gOff1 + 8192;

  v8i cg[2][2] = {};
  v8i cu[2][2] = {};

  auto compute = [&](const int8_t* L) {
    v8i af0 = load_a_frag(L + aOff0);
    v8i af1 = load_a_frag(L + aOff1);
    v8i bf0 = load_b_frag(L + gOff0);
    v8i bf1 = load_b_frag(L + gOff1);
    cg[0][0] = wmma_iu8(af0, bf0, cg[0][0]);
    cg[1][0] = wmma_iu8(af1, bf0, cg[1][0]);
    cg[0][1] = wmma_iu8(af0, bf1, cg[0][1]);
    cg[1][1] = wmma_iu8(af1, bf1, cg[1][1]);
    bf0 = load_b_frag(L + uOff0);
    bf1 = load_b_frag(L + uOff1);
    cu[0][0] = wmma_iu8(af0, bf0, cu[0][0]);
    cu[1][0] = wmma_iu8(af1, bf0, cu[1][0]);
    cu[0][1] = wmma_iu8(af0, bf1, cu[0][1]);
    cu[1][1] = wmma_iu8(af1, bf1, cu[1][1]);
  };

  const int nk2 = K / 128;
  issue(0u, 0);
  for (int s = 0; s < nk2; ++s) {
    const int k0 = s * 128;
    issue(24576u, k0 + 64);
    asm volatile("s_wait_asynccnt 3" ::: "memory");
    __syncthreads();
    compute(&lds[0][0]);
    __syncthreads();
    if (s + 1 < nk2) {
      issue(0u, k0 + 128);
      asm volatile("s_wait_asynccnt 3" ::: "memory");
    } else {
      asm volatile("s_wait_asynccnt 0" ::: "memory");
    }
    __syncthreads();
    compute(&lds[1][0]);
    __syncthreads();
  }

  const float sin   = s_in[0];
  const float dq_g  = sin * s_wg[0];
  const float dq_u  = sin * s_wu[0];
  const float inv_d = 1.0f / s_dn[0];

  #pragma unroll
  for (int i = 0; i < 2; ++i) {
    #pragma unroll
    for (int j = 0; j < 2; ++j) {
      const int col = nb + j * 16 + r;
      const float bgv = bias_g[col];
      const float buv = bias_u[col];
      #pragma unroll
      for (int v = 0; v < 8; ++v) {
        const int rw = mr + i * 16 + v + 8 * h;
        float gf = (float)cg[i][j][v] * dq_g + bgv;
        float g16 = __half2float(__float2half(gf));
        float sl = g16 / (1.0f + __expf(-g16));
        sl = __half2float(__float2half(sl));
        float uf = (float)cu[i][j][v] * dq_u + buv;
        float hid = sl * uf;
        int q = (int)rintf(hid * inv_d);
        q = q < -127 ? -127 : (q > 127 ? 127 : q);
        Hq[(size_t)rw * IDIM + col] = (int8_t)q;
      }
    }
  }
}

__global__ void __launch_bounds__(512)
gemm2_down(const int8_t* __restrict__ Aq,
           const int8_t* __restrict__ Wd,
           const float* __restrict__ bias_d,
           const float* __restrict__ s_dn,
           const float* __restrict__ s_wd,
           float* __restrict__ Out)
{
  const int K = IDIM;
  __shared__ int8_t lds[2][16384];

  const int t    = threadIdx.x;
  const int wave = t >> 5;
  const int lane = t & 31;
  const int wm = wave & 3, wn = wave >> 2;
  const int block_m = blockIdx.x * 128;
  const int block_n = blockIdx.y * 128;
  const int mr = block_m + wm * 32;
  const int nb = block_n + wn * 32;
  const int r = lane & 15, h = lane >> 4;

  const int row  = t >> 2;
  const int koff = (t & 3) * 16;
  const int8_t* gA = Aq + (size_t)(block_m + row) * K + koff;
  const int8_t* gB = Wd + (size_t)(block_n + row) * K + koff;
  const uint32_t ldsBase = (uint32_t)(uintptr_t)&lds[0][0];
  const uint32_t cA = ldsBase + (uint32_t)(t * 16);
  const uint32_t cB = cA + 8192;

  auto issue = [&](uint32_t bufbase, int kb) {
    async_b128(cA + bufbase, gA + kb);
    async_b128(cB + bufbase, gB + kb);
  };

  const uint32_t aOff0 = (uint32_t)((wm * 32 + r) * 64 + h * 16);
  const uint32_t aOff1 = aOff0 + 1024;
  const uint32_t bOff0 = (uint32_t)(8192 + (wn * 32 + r) * 64 + h * 16);
  const uint32_t bOff1 = bOff0 + 1024;

  v8i c[2][2] = {};

  auto compute = [&](const int8_t* L) {
    v8i af0 = load_a_frag(L + aOff0);
    v8i af1 = load_a_frag(L + aOff1);
    v8i bf0 = load_b_frag(L + bOff0);
    v8i bf1 = load_b_frag(L + bOff1);
    c[0][0] = wmma_iu8(af0, bf0, c[0][0]);
    c[1][0] = wmma_iu8(af1, bf0, c[1][0]);
    c[0][1] = wmma_iu8(af0, bf1, c[0][1]);
    c[1][1] = wmma_iu8(af1, bf1, c[1][1]);
  };

  const int nk2 = K / 128;
  issue(0u, 0);
  for (int s = 0; s < nk2; ++s) {
    const int k0 = s * 128;
    issue(16384u, k0 + 64);
    asm volatile("s_wait_asynccnt 2" ::: "memory");
    __syncthreads();
    compute(&lds[0][0]);
    __syncthreads();
    if (s + 1 < nk2) {
      issue(0u, k0 + 128);
      asm volatile("s_wait_asynccnt 2" ::: "memory");
    } else {
      asm volatile("s_wait_asynccnt 0" ::: "memory");
    }
    __syncthreads();
    compute(&lds[1][0]);
    __syncthreads();
  }

  const float dq = s_dn[0] * s_wd[0];
  #pragma unroll
  for (int i = 0; i < 2; ++i) {
    #pragma unroll
    for (int j = 0; j < 2; ++j) {
      const int col = nb + j * 16 + r;
      const float bv = bias_d[col];
      #pragma unroll
      for (int v = 0; v < 8; ++v) {
        const int rw = mr + i * 16 + v + 8 * h;
        Out[(size_t)rw * HDIM + col] = (float)c[i][j][v] * dq + bv;
      }
    }
  }
}

extern "C" void kernel_launch(void* const* d_in, const int* in_sizes, int n_in,
                              void* d_out, int out_size, void* d_ws, size_t ws_size,
                              hipStream_t stream) {
  (void)in_sizes; (void)n_in; (void)out_size; (void)ws_size;

  const float* x             = (const float*)d_in[0];
  const int*   w_gate        = (const int*)d_in[1];
  const int*   w_up          = (const int*)d_in[2];
  const int*   w_down        = (const int*)d_in[3];
  const float* b_gate        = (const float*)d_in[4];
  const float* b_up          = (const float*)d_in[5];
  const float* b_down        = (const float*)d_in[6];
  const float* gate_in_scale = (const float*)d_in[7];
  const float* down_in_scale = (const float*)d_in[8];
  const float* w_gate_scale  = (const float*)d_in[9];
  const float* w_up_scale    = (const float*)d_in[10];
  const float* w_down_scale  = (const float*)d_in[11];

  uint8_t* ws = (uint8_t*)d_ws;
  size_t off = 0;
  int8_t* xq  = (int8_t*)(ws + off); off += (size_t)M_TOK * HDIM;
  int8_t* wg8 = (int8_t*)(ws + off); off += (size_t)IDIM * HDIM;
  int8_t* wu8 = (int8_t*)(ws + off); off += (size_t)IDIM * HDIM;
  int8_t* wd8 = (int8_t*)(ws + off); off += (size_t)HDIM * IDIM;
  int8_t* hq  = (int8_t*)(ws + off); off += (size_t)M_TOK * IDIM;

  {
    int n4 = (IDIM * HDIM) / 4;
    dim3 grid(n4 / 256);
    pack_w_i8<<<grid, 256, 0, stream>>>(w_gate, (uint32_t*)wg8, n4);
    pack_w_i8<<<grid, 256, 0, stream>>>(w_up,   (uint32_t*)wu8, n4);
    pack_w_i8<<<grid, 256, 0, stream>>>(w_down, (uint32_t*)wd8, n4);
  }
  {
    int n4 = (M_TOK * HDIM) / 4;
    quant_x_i8<<<dim3(n4 / 256), 256, 0, stream>>>(x, (uint32_t*)xq, gate_in_scale, n4);
  }
  {
    dim3 grid(M_TOK / 128, IDIM / 128);
    gemm1_gate_up_silu<<<grid, 512, 0, stream>>>(
        xq, wg8, wu8, b_gate, b_up,
        gate_in_scale, w_gate_scale, w_up_scale, down_in_scale, hq);
  }
  {
    dim3 grid(M_TOK / 128, HDIM / 128);
    gemm2_down<<<grid, 512, 0, stream>>>(
        hq, wd8, b_down, down_in_scale, w_down_scale, (float*)d_out);
  }
}
